// TransformerBlock_M_17755394801893
// MI455X (gfx1250) — compile-verified, matrix operand order checked by reading, not run on hardware
//
#include <hip/hip_runtime.h>
#include <hip/hip_bf16.h>
#include <float.h>

// ---------------------------------------------------------------------------
// Point-transformer block for MI455X (gfx1250, wave32, WMMA).
// All matmuls use v_wmma_f32_16x16x32_f16 (f16 operands, f32 accumulate).
// Round 2: register-blocked GEMMs (A-fragments preloaded per M-tile, swept
// across all N-tiles), B-fragment reuse in the fused pos-GEMM.
// ---------------------------------------------------------------------------

#define BATCH 4
#define NPTS  4096
#define KNN_K 16
#define DP    128
#define DM    256
#define TOTPT (BATCH * NPTS)   // 16384

typedef __attribute__((ext_vector_type(16))) _Float16 v16h;
typedef __attribute__((ext_vector_type(8)))  float    v8f;

// ---- WMMA helpers ----------------------------------------------------------

__device__ __forceinline__ v8f wmma_f16(v16h a, v16h b, v8f c) {
  // 8 args: (neg_a, A, neg_b, B, c_mod, C, reuse_a, reuse_b)
  return __builtin_amdgcn_wmma_f32_16x16x32_f16(
      false, a, false, b, (short)0, c, false, false);
}

// Load a 16x32 f16 fragment whose K-dimension is contiguous in memory.
// rowptr must point at element [row][kbase] of a row-major (K-contiguous)
// matrix, where row = tile_base + (lane & 15).
// Layout per ISA 7.12.2 (16-bit A-matrix 16x32):
//   lane half h = lane>>4 ; dword v (0..7) holds K = (v<4?0:16) + 8h + 2(v&3), +1
__device__ __forceinline__ v16h load_frag16(const _Float16* rowptr) {
  const int lane = (int)(threadIdx.x & 31u);
  const int h = lane >> 4;
  union { v16h v; unsigned u[8]; } r;
#pragma unroll
  for (int e = 0; e < 8; ++e) {
    const int k = ((e < 4) ? 0 : 16) + h * 8 + (e & 3) * 2;
    r.u[e] = *(const unsigned*)(rowptr + k);
  }
  return r.v;
}

// ---- Kernel 1: brute-force KNN (top-16 by squared distance, self included) -

__global__ __launch_bounds__(256) void knn_kernel(const float* __restrict__ xyz,
                                                  int* __restrict__ knn_idx) {
  __shared__ float sx[256], sy[256], sz[256];
  const int b = blockIdx.y;
  const int n = blockIdx.x * 256 + threadIdx.x;
  const float* base = xyz + (size_t)b * NPTS * 3;
  const float qx = base[n * 3 + 0];
  const float qy = base[n * 3 + 1];
  const float qz = base[n * 3 + 2];

  float dist[KNN_K];
  int   idx[KNN_K];
#pragma unroll
  for (int i = 0; i < KNN_K; ++i) { dist[i] = FLT_MAX; idx[i] = 0; }

  for (int tile = 0; tile < NPTS; tile += 256) {
    const int j = tile + threadIdx.x;
    sx[threadIdx.x] = base[j * 3 + 0];
    sy[threadIdx.x] = base[j * 3 + 1];
    sz[threadIdx.x] = base[j * 3 + 2];
    __syncthreads();
    for (int jj = 0; jj < 256; ++jj) {
      const float dx = qx - sx[jj];
      const float dy = qy - sy[jj];
      const float dz = qz - sz[jj];
      const float d2 = dx * dx + dy * dy + dz * dz;
      if (d2 < dist[KNN_K - 1]) {
        dist[KNN_K - 1] = d2;
        idx[KNN_K - 1] = tile + jj;
#pragma unroll
        for (int s = KNN_K - 1; s > 0; --s) {
          if (dist[s] < dist[s - 1]) {
            float td = dist[s]; dist[s] = dist[s - 1]; dist[s - 1] = td;
            int   ti = idx[s];  idx[s]  = idx[s - 1];  idx[s - 1]  = ti;
          }
        }
      }
    }
    __syncthreads();
  }

  int* out = knn_idx + ((size_t)b * NPTS + n) * KNN_K;
#pragma unroll
  for (int i = 0; i < KNN_K; ++i) out[i] = idx[i];
}

// ---- Kernel 2: f32 -> f16 convert (plain / transposed) ---------------------

__global__ void conv_f16_kernel(const float* __restrict__ src,
                                _Float16* __restrict__ dst, int n) {
  const int i = blockIdx.x * 256 + threadIdx.x;
  if (i < n) dst[i] = (_Float16)src[i];
}

// dst[c][r] = src[r][c]; src is rows x cols row-major; dst is cols x rows.
__global__ void convT_f16_kernel(const float* __restrict__ src,
                                 _Float16* __restrict__ dst, int rows, int cols) {
  const int i = blockIdx.x * 256 + threadIdx.x;
  if (i < rows * cols) {
    const int r = i / cols, c = i % cols;
    dst[(size_t)c * rows + r] = (_Float16)src[i];
  }
}

// ---- Kernel 3: register-blocked WMMA GEMM ----------------------------------
// Out[M,NW] = A[M,KD] * BT[NW,KD]^T (+ bias). One wave per 16-row M-tile;
// the wave preloads all KD/32 A-fragments into registers and sweeps every
// N-tile, so A is read exactly once.

template <int KD, int NW, bool HAS_BIAS>
__global__ __launch_bounds__(32) void gemm16n_kernel(
    const _Float16* __restrict__ A, const _Float16* __restrict__ BT,
    const float* __restrict__ bias, _Float16* __restrict__ Out) {
  constexpr int KSTEPS = KD / 32;
  const int lane = (int)threadIdx.x;
  const int h = lane >> 4, l16 = lane & 15;
  const int mbase = blockIdx.x * 16;

  const _Float16* arow = A + (size_t)(mbase + l16) * KD;
  v16h af[KSTEPS];
#pragma unroll
  for (int ks = 0; ks < KSTEPS; ++ks) af[ks] = load_frag16(arow + ks * 32);

  for (int nt = 0; nt < NW / 16; ++nt) {
    const _Float16* brow = BT + (size_t)(nt * 16 + l16) * KD;
    v8f acc = {};
#pragma unroll
    for (int ks = 0; ks < KSTEPS; ++ks)
      acc = wmma_f16(af[ks], load_frag16(brow + ks * 32), acc);

    const int col = nt * 16 + l16;
    const float bv = HAS_BIAS ? bias[col] : 0.0f;
    _Float16* orow = Out + (size_t)(mbase + h * 8) * NW + col;
#pragma unroll
    for (int r = 0; r < 8; ++r)
      orow[(size_t)r * NW] = (_Float16)(acc[r] + bv);
  }
}

// ---- Kernel 4: fused pos-MLP + attention -----------------------------------
// One workgroup = 8 query points (128 rows of the pos GEMM), 128 threads.

#define NPER 8  // points per workgroup; rows = NPER*KNN_K = 128

__global__ __launch_bounds__(128) void attn_kernel(
    const float* __restrict__ xyz, const int* __restrict__ knn,
    const float* __restrict__ fd1_w, const float* __restrict__ fd1_b,
    const _Float16* __restrict__ fd2T, const float* __restrict__ fd2_b,
    const _Float16* __restrict__ Q16, const _Float16* __restrict__ K16,
    const _Float16* __restrict__ V16, _Float16* __restrict__ R16) {
  extern __shared__ char smem[];
  _Float16* Hs = (_Float16*)smem;          // [128][256] relu(delta*fd1+b1)
  _Float16* Ps = Hs + 128 * 256;           // [128][256] pos (f16)
  float* w1 = (float*)(Ps + 128 * 256);    // [3*256]
  float* b1 = w1 + 3 * 256;                // [256]
  float* lg = b1 + 256;                    // [8][16] logits
  float* aw = lg + 128;                    // [8][16] softmax weights
  int* nrow = (int*)(aw + 128);            // [128] global neighbor rows

  const int t = (int)threadIdx.x;
  const int p0 = blockIdx.x * NPER;        // global point base (never crosses b)
  const int b = p0 >> 12;                  // / NPTS

  for (int i = t; i < 3 * 256; i += 128) w1[i] = fd1_w[i];
  for (int i = t; i < 256; i += 128)     b1[i] = fd1_b[i];

  // phase 1: H = relu(delta @ fd1_w + fd1_b), one row per thread
  const int p = t >> 4, kk = t & 15;
  const int pt = p0 + p;
  {
    const int nb = knn[(size_t)pt * KNN_K + kk];
    const int grow = (b << 12) + nb;
    nrow[t] = grow;
    const float dx = xyz[pt * 3 + 0] - xyz[grow * 3 + 0];
    const float dy = xyz[pt * 3 + 1] - xyz[grow * 3 + 1];
    const float dz = xyz[pt * 3 + 2] - xyz[grow * 3 + 2];
    __syncthreads();  // w1/b1 ready
    _Float16* hrow = Hs + t * 256;
    for (int c = 0; c < 256; ++c) {
      const float v = dx * w1[c] + dy * w1[256 + c] + dz * w1[512 + c] + b1[c];
      hrow[c] = (_Float16)(v > 0.0f ? v : 0.0f);
    }
  }
  __syncthreads();

  // phase 2: Ps = Hs @ fd2T^T + fd2_b  (128x256x256 GEMM, WMMA).
  // B-fragments (global fd2T) preloaded per N-tile and reused across the
  // wave's two M-tiles; A-fragments re-read from LDS (cheap, WGP-local).
  {
    const int w = t >> 5, lane = t & 31, h = lane >> 4, l16 = lane & 15;
    for (int nt = 0; nt < 16; ++nt) {
      const _Float16* brow = fd2T + (size_t)(nt * 16 + l16) * 256;
      v16h bf[8];
#pragma unroll
      for (int ks = 0; ks < 8; ++ks) bf[ks] = load_frag16(brow + ks * 32);

      const int col = nt * 16 + l16;
      const float bv = fd2_b[col];
#pragma unroll
      for (int mi = 0; mi < 2; ++mi) {
        const int mt = w * 2 + mi;
        const _Float16* arow = Hs + (size_t)(mt * 16 + l16) * 256;
        v8f acc = {};
#pragma unroll
        for (int ks = 0; ks < 8; ++ks)
          acc = wmma_f16(load_frag16(arow + ks * 32), bf[ks], acc);
#pragma unroll
        for (int r = 0; r < 8; ++r)
          Ps[(size_t)(mt * 16 + h * 8 + r) * 256 + col] = (_Float16)(acc[r] + bv);
      }
    }
  }
  __syncthreads();

  // phase 3: logits[p][k] = q . (kf + pos) / sqrt(DM)
  {
    const _Float16* q  = Q16 + (size_t)pt * 256;
    const _Float16* kf = K16 + (size_t)nrow[t] * 256;
    const _Float16* pp = Ps + (size_t)t * 256;
    float s = 0.0f;
    for (int d = 0; d < 256; ++d)
      s += (float)q[d] * ((float)kf[d] + (float)pp[d]);
    lg[t] = s * (1.0f / 16.0f);
  }
  __syncthreads();

  if (t < NPER) {
    float m = -1e30f;
    for (int k = 0; k < KNN_K; ++k) m = fmaxf(m, lg[t * 16 + k]);
    float e[KNN_K], s = 0.0f;
    for (int k = 0; k < KNN_K; ++k) { e[k] = __expf(lg[t * 16 + k] - m); s += e[k]; }
    const float inv = 1.0f / s;
    for (int k = 0; k < KNN_K; ++k) aw[t * 16 + k] = e[k] * inv;
  }
  __syncthreads();

  // phase 4: res[p][d] = sum_k attn[p][k] * (vf + pos); 16 dims per thread
  {
    const int db = (t & 15) * 16;
    _Float16* outp = R16 + (size_t)pt * 256 + db;
    float acc[16];
#pragma unroll
    for (int i = 0; i < 16; ++i) acc[i] = 0.0f;
    for (int k = 0; k < KNN_K; ++k) {
      const float a = aw[p * 16 + k];
      const _Float16* vf = V16 + (size_t)nrow[p * 16 + k] * 256 + db;
      const _Float16* pp = Ps + (size_t)(p * 16 + k) * 256 + db;
#pragma unroll
      for (int i = 0; i < 16; ++i)
        acc[i] += a * ((float)vf[i] + (float)pp[i]);
    }
#pragma unroll
    for (int i = 0; i < 16; ++i) outp[i] = (_Float16)acc[i];
  }
}

// ---- Kernel 5: out = res @ fc2_w + fc2_b + features, transposed to [B,DP,N] -
// Same register-blocked structure: one wave per M-tile, A preloaded, sweeps
// all 8 N-tiles, fuses bias + residual + [B,DP,N] transpose.

__global__ __launch_bounds__(32) void outproj_kernel(
    const _Float16* __restrict__ R16, const _Float16* __restrict__ fc2T,
    const float* __restrict__ fc2_b, const float* __restrict__ features,
    float* __restrict__ out) {
  const int lane = (int)threadIdx.x;
  const int h = lane >> 4, l16 = lane & 15;
  const int mbase = blockIdx.x * 16;   // rows (b*N+n)

  const _Float16* arow = R16 + (size_t)(mbase + l16) * DM;
  v16h af[8];
#pragma unroll
  for (int ks = 0; ks < 8; ++ks) af[ks] = load_frag16(arow + ks * 32);

  for (int nt = 0; nt < DP / 16; ++nt) {
    const _Float16* brow = fc2T + (size_t)(nt * 16 + l16) * DM;
    v8f acc = {};
#pragma unroll
    for (int ks = 0; ks < 8; ++ks)
      acc = wmma_f16(af[ks], load_frag16(brow + ks * 32), acc);

    const int col = nt * 16 + l16;
    const float bv = fc2_b[col];
#pragma unroll
    for (int r = 0; r < 8; ++r) {
      const int row = mbase + h * 8 + r;
      const int bb = row >> 12, n = row & 4095;
      out[((size_t)(bb * DP + col) << 12) + n] =
          acc[r] + bv + features[(size_t)row * DP + col];
    }
  }
}

// ---------------------------------------------------------------------------

extern "C" void kernel_launch(void* const* d_in, const int* in_sizes, int n_in,
                              void* d_out, int out_size, void* d_ws, size_t ws_size,
                              hipStream_t stream) {
  (void)in_sizes; (void)n_in; (void)out_size; (void)ws_size;
  const float* features = (const float*)d_in[0];
  const float* xyz      = (const float*)d_in[1];
  const float* fc1_w    = (const float*)d_in[2];
  const float* fc1_b    = (const float*)d_in[3];
  const float* fc2_w    = (const float*)d_in[4];
  const float* fc2_b    = (const float*)d_in[5];
  const float* fd1_w    = (const float*)d_in[6];
  const float* fd1_b    = (const float*)d_in[7];
  const float* fd2_w    = (const float*)d_in[8];
  const float* fd2_b    = (const float*)d_in[9];
  const float* wq       = (const float*)d_in[10];
  const float* wk       = (const float*)d_in[11];
  const float* wv       = (const float*)d_in[12];
  float* out = (float*)d_out;

  // ---- carve workspace ----
  char* ws = (char*)d_ws;
  size_t off = 0;
  auto carve = [&](size_t bytes) -> char* {
    char* p = ws + off;
    off += (bytes + 255) & ~(size_t)255;
    return p;
  };
  int*       knn_idx = (int*)      carve((size_t)TOTPT * KNN_K * 4);
  _Float16*  feat16  = (_Float16*) carve((size_t)TOTPT * DP * 2);
  _Float16*  x16     = (_Float16*) carve((size_t)TOTPT * DM * 2);
  _Float16*  q16     = (_Float16*) carve((size_t)TOTPT * DM * 2);
  _Float16*  k16     = (_Float16*) carve((size_t)TOTPT * DM * 2);
  _Float16*  v16     = (_Float16*) carve((size_t)TOTPT * DM * 2);
  _Float16*  r16     = (_Float16*) carve((size_t)TOTPT * DM * 2);
  _Float16*  fc1T    = (_Float16*) carve((size_t)DM * DP * 2);   // [DM][DP]
  _Float16*  wqT     = (_Float16*) carve((size_t)DM * DM * 2);
  _Float16*  wkT     = (_Float16*) carve((size_t)DM * DM * 2);
  _Float16*  wvT     = (_Float16*) carve((size_t)DM * DM * 2);
  _Float16*  fd2T    = (_Float16*) carve((size_t)DM * DM * 2);
  _Float16*  fc2T    = (_Float16*) carve((size_t)DP * DM * 2);   // [DP][DM]

  // ---- 1. KNN ----
  knn_kernel<<<dim3(NPTS / 256, BATCH), 256, 0, stream>>>(xyz, knn_idx);

  // ---- 2. converts ----
  {
    const int nf = TOTPT * DP;
    conv_f16_kernel<<<(nf + 255) / 256, 256, 0, stream>>>(features, feat16, nf);
    convT_f16_kernel<<<(DP * DM + 255) / 256, 256, 0, stream>>>(fc1_w, fc1T, DP, DM);
    convT_f16_kernel<<<(DM * DM + 255) / 256, 256, 0, stream>>>(wq, wqT, DM, DM);
    convT_f16_kernel<<<(DM * DM + 255) / 256, 256, 0, stream>>>(wk, wkT, DM, DM);
    convT_f16_kernel<<<(DM * DM + 255) / 256, 256, 0, stream>>>(wv, wvT, DM, DM);
    convT_f16_kernel<<<(DM * DM + 255) / 256, 256, 0, stream>>>(fd2_w, fd2T, DM, DM);
    convT_f16_kernel<<<(DM * DP + 255) / 256, 256, 0, stream>>>(fc2_w, fc2T, DM, DP);
  }

  // ---- 3. x = features @ fc1 + b ----
  gemm16n_kernel<DP, DM, true><<<TOTPT / 16, 32, 0, stream>>>(
      feat16, fc1T, fc1_b, x16);

  // ---- 4. q/k/v projections ----
  gemm16n_kernel<DM, DM, false><<<TOTPT / 16, 32, 0, stream>>>(
      x16, wqT, nullptr, q16);
  gemm16n_kernel<DM, DM, false><<<TOTPT / 16, 32, 0, stream>>>(
      x16, wkT, nullptr, k16);
  gemm16n_kernel<DM, DM, false><<<TOTPT / 16, 32, 0, stream>>>(
      x16, wvT, nullptr, v16);

  // ---- 5. fused pos-MLP + attention ----
  {
    const size_t smem = (size_t)128 * 256 * 2 * 2   // Hs + Ps (f16)
                      + (3 * 256 + 256 + 128 + 128) * 4  // w1,b1,lg,aw
                      + 128 * 4;                    // nrow
    attn_kernel<<<TOTPT / NPER, 128, smem, stream>>>(
        xyz, knn_idx, fd1_w, fd1_b, fd2T, fd2_b, q16, k16, v16, r16);
  }

  // ---- 6. output projection + residual + transpose ----
  outproj_kernel<<<TOTPT / 16, 32, 0, stream>>>(
      r16, fc2T, fc2_b, features, out);
}
